// UpdateNet_29867202576661
// MI455X (gfx1250) — compile-verified, matrix operand order checked by reading, not run on hardware
//
#include <hip/hip_runtime.h>

typedef __attribute__((ext_vector_type(16))) __bf16 v16bf;
typedef __attribute__((ext_vector_type(8)))  __bf16 v8bf;
typedef __attribute__((ext_vector_type(8)))  float  v8f;
typedef __attribute__((ext_vector_type(4)))  float  v4f;

#define C_IN   16
#define K1PAD  64     // 3*C = 48 padded to 64 (two K=32 WMMA tiles)
#define NHID   128
#define HDIM   256
#define WDIM   256
#define HW     (HDIM * WDIM)
#define WAVES  8      // waves per block (wave32)
#define NBLK   2048   // persistent-ish grid; each wave strides over strips

// Load one bf16 WMMA fragment K=32 slice from LDS: a lane holds 8 contiguous
// K values and the 8 values 16 further on (two aligned 16-byte chunks).
__device__ __forceinline__ v16bf lds_frag(const __bf16* rowptr) {
    v8bf lo = *(const v8bf*)(rowptr);
    v8bf hi = *(const v8bf*)(rowptr + 16);
    return __builtin_shufflevector(lo, hi, 0,1,2,3,4,5,6,7,8,9,10,11,12,13,14,15);
}

// Build x / sobel_x / sobel_y features for this lane's 8 channels.
// INTERIOR=true: no bounds checks, 9 plain b32 loads per channel off one base
// pointer (offsets fold into the instruction's immediate offset).
template <bool INTERIOR>
__device__ __forceinline__ void build_features(
        const float* __restrict__ xb, __bf16 (*feat)[K1PAD],
        int p, int grp, int hy, int wp) {
    #pragma unroll
    for (int c8 = 0; c8 < 8; ++c8) {
        const int c = grp * 8 + c8;              // lanes 0-15: c 0..7, 16-31: 8..15
        const float* xc = xb + (size_t)c * HW;
        float v[3][3];
        if (INTERIOR) {
            const float* pr = xc + (hy - 1) * WDIM + (wp - 1);
            #pragma unroll
            for (int dh = 0; dh < 3; ++dh)
                #pragma unroll
                for (int dw = 0; dw < 3; ++dw)
                    v[dh][dw] = pr[dh * WDIM + dw];
        } else {
            #pragma unroll
            for (int dh = 0; dh < 3; ++dh) {
                const int yy = hy + dh - 1;
                const bool hok = (unsigned)yy < HDIM;
                #pragma unroll
                for (int dw = 0; dw < 3; ++dw) {
                    const int xx = wp + dw - 1;
                    v[dh][dw] = (hok && (unsigned)xx < WDIM) ? xc[yy * WDIM + xx] : 0.0f;
                }
            }
        }
        const float gx = (v[0][2] - v[0][0]) + 2.0f * (v[1][2] - v[1][0]) + (v[2][2] - v[2][0]);
        const float gy = (v[2][0] - v[0][0]) + 2.0f * (v[2][1] - v[0][1]) + (v[2][2] - v[0][2]);
        feat[p][c]      = (__bf16)v[1][1];
        feat[p][16 + c] = (__bf16)gx;
        feat[p][32 + c] = (__bf16)gy;
    }
}

__global__ __launch_bounds__(256) void updatenet_wmma_bf16(
        const float* __restrict__ x,     // [B,16,256,256]
        const float* __restrict__ w1,    // [128,48]
        const float* __restrict__ w2,    // [16,128]
        float* __restrict__ out,         // [B,16,256,256]
        int total_strips) {

    __shared__ __bf16 w1s[NHID][K1PAD];          // 16 KB, K-padded bf16 weights
    __shared__ __bf16 w2s[C_IN][NHID];           //  4 KB
    __shared__ __bf16 feats[WAVES][16][K1PAD];   // 16 KB, per-wave feature tile
    __shared__ __bf16 hs[WAVES][16][NHID];       // 32 KB, per-wave hidden tile

    // ---- Stage weights fp32 -> bf16 into LDS once per block ----
    for (int i = threadIdx.x; i < NHID * K1PAD; i += 256) {
        int n = i >> 6, k = i & (K1PAD - 1);
        w1s[n][k] = (k < 48) ? (__bf16)w1[n * 48 + k] : (__bf16)0.0f;
    }
    for (int i = threadIdx.x; i < C_IN * NHID; i += 256) {
        w2s[i >> 7][i & (NHID - 1)] = (__bf16)w2[i];
    }
    __syncthreads();

    const int lane = threadIdx.x & 31;
    const int warp = threadIdx.x >> 5;
    const int p    = lane & 15;   // pixel-in-strip (M) / N-column role
    const int grp  = lane >> 4;   // which K half-group this lane carries
    const int kb   = grp * 8;     // lane's K base inside a K=32 tile

    __bf16 (*feat)[K1PAD] = feats[warp];
    __bf16 (*hrow)[NHID]  = hs[warp];

    // K padding columns 48..63 stay zero for every strip: write once.
    #pragma unroll
    for (int i = 0; i < 8; ++i) feat[p][48 + kb + i] = (__bf16)0.0f;

    // Strip-invariant w2 B fragments (GEMM2 then only loads A from LDS).
    const v16bf wb0 = lds_frag(&w2s[p][kb]);
    const v16bf wb1 = lds_frag(&w2s[p][kb + 32]);
    const v16bf wb2 = lds_frag(&w2s[p][kb + 64]);
    const v16bf wb3 = lds_frag(&w2s[p][kb + 96]);

    // ---- Grid-stride loop over 16-pixel strips ----
    for (int s = blockIdx.x * WAVES + warp; s < total_strips; s += NBLK * WAVES) {
        const int w0 = (s & 15) << 4;
        const int hy = (s >> 4) & (HDIM - 1);
        const int b  = s >> 12;
        const int wp = w0 + p;

        const float* xb = x + (size_t)b * C_IN * HW;
        // wave-uniform branch: all 16 pixels interior (hy 1..254, w0 16..224)
        const bool interior = (hy - 1u < (unsigned)(HDIM - 2)) &
                              (w0 - 16u < (unsigned)(WDIM - 32));
        if (interior) build_features<true >(xb, feat, p, grp, hy, wp);
        else          build_features<false>(xb, feat, p, grp, hy, wp);
        // per-wave LDS ops are in-order; no block barrier needed

        // ---- GEMM1: h[16px][128] = feat[16][64] x w1^T (K = 2x32) ----
        const v16bf a0 = lds_frag(&feat[p][kb]);
        const v16bf a1 = lds_frag(&feat[p][kb + 32]);

        #pragma unroll
        for (int n = 0; n < 8; ++n) {
            v8f acc = {};
            const v16bf b0 = lds_frag(&w1s[n * 16 + p][kb]);
            const v16bf b1 = lds_frag(&w1s[n * 16 + p][kb + 32]);
            acc = __builtin_amdgcn_wmma_f32_16x16x32_bf16(false, a0, false, b0,
                                                          (short)0, acc, false, false);
            acc = __builtin_amdgcn_wmma_f32_16x16x32_bf16(false, a1, false, b1,
                                                          (short)0, acc, false, false);
            // D layout: lane holds col n*16+p, rows m = r + 8*grp. ReLU->bf16->LDS
            #pragma unroll
            for (int r = 0; r < 8; ++r) {
                float hv = acc[r];
                hv = hv > 0.0f ? hv : 0.0f;
                hrow[r + grp * 8][n * 16 + p] = (__bf16)hv;
            }
        }

        // ---- GEMM2: update[16px][16ch] = h[16][128] x w2^T (K = 4x32) ----
        v8f acc2 = {};
        acc2 = __builtin_amdgcn_wmma_f32_16x16x32_bf16(false, lds_frag(&hrow[p][kb]),
                                                       false, wb0, (short)0, acc2, false, false);
        acc2 = __builtin_amdgcn_wmma_f32_16x16x32_bf16(false, lds_frag(&hrow[p][kb + 32]),
                                                       false, wb1, (short)0, acc2, false, false);
        acc2 = __builtin_amdgcn_wmma_f32_16x16x32_bf16(false, lds_frag(&hrow[p][kb + 64]),
                                                       false, wb2, (short)0, acc2, false, false);
        acc2 = __builtin_amdgcn_wmma_f32_16x16x32_bf16(false, lds_frag(&hrow[p][kb + 96]),
                                                       false, wb3, (short)0, acc2, false, false);

        // ---- Store: lane holds channel p, pixels w0 + 8*grp + r (contiguous)
        float* op = out + ((size_t)b * C_IN + p) * HW + (size_t)hy * WDIM + w0 + grp * 8;
        const v4f lo = __builtin_shufflevector(acc2, acc2, 0, 1, 2, 3);
        const v4f hi = __builtin_shufflevector(acc2, acc2, 4, 5, 6, 7);
        *(v4f*)(op)     = lo;
        *(v4f*)(op + 4) = hi;
    }
}

extern "C" void kernel_launch(void* const* d_in, const int* in_sizes, int n_in,
                              void* d_out, int out_size, void* d_ws, size_t ws_size,
                              hipStream_t stream) {
    const float* x  = (const float*)d_in[0];
    const float* w1 = (const float*)d_in[1];
    const float* w2 = (const float*)d_in[2];
    float* out = (float*)d_out;

    const int B = in_sizes[0] / (C_IN * HW);            // 32
    const int strips = B * HDIM * (WDIM / 16);          // 131072

    updatenet_wmma_bf16<<<NBLK, 256, 0, stream>>>(x, w1, w2, out, strips);
}
